// MultiHeadAttention_73650099191976
// MI455X (gfx1250) — matrix operand order repaired and checked by reading, not compiled, not run on hardware
//
#include <hip/hip_runtime.h>
#include <math.h>

typedef __attribute__((ext_vector_type(16))) _Float16 v16h;
typedef __attribute__((ext_vector_type(8)))  _Float16 v8h;
typedef __attribute__((ext_vector_type(4)))  _Float16 v4h;
typedef __attribute__((ext_vector_type(8)))  float    v8f;

#define D_MODEL 1024
#define N_HEADS 16
#define D_HEAD  64
#define BATCH   4
#define SEQ     2048
#define M_ROWS  (BATCH * SEQ)          // 8192 rows for the big GEMMs

static __device__ __forceinline__ v8h ld8h(const _Float16* p) {
  return *(const v8h*)p;               // 16B-aligned contiguous 8 halfs
}

static __device__ __forceinline__ v8f wmma_f16(v16h a, v16h b, v8f c) {
  // D = A(16x32 f16) * B(32x16 f16) + C(16x16 f32)
  return __builtin_amdgcn_wmma_f32_16x16x32_f16(false, a, false, b, (short)0, c,
                                                false, false);
}

// A-fragment (16x32 f16) from an fp32 row-major source; p already includes
// the per-lane row (M=lane&15) and kb = 8*(lane>=16) offsets.
static __device__ __forceinline__ v16h ldA_f32(const float* p) {
  const float4* pa = (const float4*)p;
  const float4 f0 = pa[0], f1 = pa[1];   // K = kb .. kb+7
  const float4 f2 = pa[4], f3 = pa[5];   // K = kb+16 .. kb+23
  v16h a;
  a[0]=(_Float16)f0.x;  a[1]=(_Float16)f0.y;  a[2]=(_Float16)f0.z;  a[3]=(_Float16)f0.w;
  a[4]=(_Float16)f1.x;  a[5]=(_Float16)f1.y;  a[6]=(_Float16)f1.z;  a[7]=(_Float16)f1.w;
  a[8]=(_Float16)f2.x;  a[9]=(_Float16)f2.y;  a[10]=(_Float16)f2.z; a[11]=(_Float16)f2.w;
  a[12]=(_Float16)f3.x; a[13]=(_Float16)f3.y; a[14]=(_Float16)f3.z; a[15]=(_Float16)f3.w;
  return a;
}

// A-fragment from an f16 row-major source (chunks at +0 and +16 halfs).
static __device__ __forceinline__ v16h ldA_f16(const _Float16* p) {
  const v8h a0 = ld8h(p), a1 = ld8h(p + 16);
  v16h a;
  #pragma unroll
  for (int j = 0; j < 8; ++j) { a[j] = a0[j]; a[8 + j] = a1[j]; }
  return a;
}

// B-fragment (32x16 f16): chunks at +0 and +16 halfs (p includes kb = 8*hi).
static __device__ __forceinline__ v16h ldB_f16(const _Float16* p) {
  const v8h b0 = ld8h(p), b1 = ld8h(p + 16);
  v16h b;
  #pragma unroll
  for (int j = 0; j < 8; ++j) { b[j] = b0[j]; b[8 + j] = b1[j]; }
  return b;
}

// ---------------------------------------------------------------------------
// fp32 -> f16 weight downcast (wq/wk/wv are [H,dh,D] == row n=h*64+e, col d)
// ---------------------------------------------------------------------------
__global__ __launch_bounds__(256) void cvt_f32_f16(const float* __restrict__ s,
                                                   _Float16* __restrict__ d) {
  const int i = blockIdx.x * 256 + threadIdx.x;   // 262144 threads x4 = 1M elems
  const float4 f = ((const float4*)s)[i];
  v4h h = {(_Float16)f.x, (_Float16)f.y, (_Float16)f.z, (_Float16)f.w};
  *(v4h*)(d + (size_t)i * 4) = h;
}

// ---------------------------------------------------------------------------
// QKV projection. One wave -> 32(M) x 64(N) output tile (one head's columns):
// 2 A-frags x 4 B-frags -> 8 WMMAs per 32-wide K step.
// VT=false: store [B,H,S,dh] f16 (Q,K).  VT=true: store [B,H,dh,S] f16 (V^T).
// ---------------------------------------------------------------------------
template <bool VT>
__global__ __launch_bounds__(128) void proj_qkv(const float* __restrict__ X,
                                                const _Float16* __restrict__ W,
                                                const float* __restrict__ bias,
                                                _Float16* __restrict__ Out,
                                                float scale) {
  const int lane = threadIdx.x & 31;
  const int wave = threadIdx.x >> 5;
  const int tile = blockIdx.x * 4 + wave;        // 0..4095
  const int tm   = tile >> 4;                    // 0..255  (32-row blocks)
  const int tng  = tile & 15;                    // 0..15   (64-col group == head)
  const int m    = lane & 15;
  const int hi   = lane >> 4;

  const float*    arow0 = X + (size_t)(tm * 32 + m) * D_MODEL + hi * 8;
  const float*    arow1 = arow0 + (size_t)16 * D_MODEL;
  const _Float16* brow  = W + (size_t)(tng * 64 + m) * D_MODEL + hi * 8;

  v8f acc[2][4] = {};
  #pragma unroll 2
  for (int k0 = 0; k0 < D_MODEL; k0 += 32) {
    const v16h a0 = ldA_f32(arow0 + k0);
    const v16h a1 = ldA_f32(arow1 + k0);
    #pragma unroll
    for (int nt = 0; nt < 4; ++nt) {
      const v16h b = ldB_f16(brow + (size_t)nt * 16 * D_MODEL + k0);
      acc[0][nt] = wmma_f16(a0, b, acc[0][nt]);
      acc[1][nt] = wmma_f16(a1, b, acc[1][nt]);
    }
  }

  const int bidx = tm >> 6;                      // batch (32-row tile never splits)
  const int s0   = (tm * 32) & (SEQ - 1);
  const int bh   = bidx * N_HEADS + tng;
  #pragma unroll
  for (int nt = 0; nt < 4; ++nt) {
    const int e = nt * 16 + m;                   // dh column
    const float bb = bias[tng * 64 + e];
    #pragma unroll
    for (int mi = 0; mi < 2; ++mi) {
      if (VT) {
        v8h hv;                                  // 8 consecutive s positions
        #pragma unroll
        for (int r = 0; r < 8; ++r) hv[r] = (_Float16)((acc[mi][nt][r] + bb) * scale);
        *(v8h*)(Out + ((size_t)bh * 64 + e) * SEQ + s0 + mi * 16 + hi * 8) = hv;
      } else {
        #pragma unroll
        for (int r = 0; r < 8; ++r) {
          const int s = s0 + mi * 16 + r + hi * 8;
          Out[((size_t)(bh * SEQ + s) << 6) + e] =
              (_Float16)((acc[mi][nt][r] + bb) * scale);
        }
      }
    }
  }
}

// ---------------------------------------------------------------------------
// One 16(q) x 32(k) flash-attention step.  MASKED=true only for the diagonal.
// Raw f32 scores are bounced through LDS into A-layout, so each lane holds 16
// scores of ONE row: row max is a local VALU tree + one shfl_xor(16); exp()
// builds the f16 P fragment in place (no second LDS bounce).  Row-sums come
// from an extra WMMA against an all-ones B fragment.  The per-row rescale
// factor is broadcast back to C-layout via a 16-float LDS array (rows
// hi*8..hi*8+7 are contiguous -> 2 ds_load_b128).
// ---------------------------------------------------------------------------
template <bool MASKED>
static __device__ __forceinline__ void attn_block(
    int kg, int qrow0, int m, int hi,
    const _Float16* __restrict__ Kh, const _Float16* __restrict__ Vth,
    float* __restrict__ sl,            // per-wave LDS: 512 scores + 16 corr
    const v16h& aq0, const v16h& aq1, const v16h& bones,
    float& mrow, v8f& lacc, v8f (&oacc)[4]) {
  // scores for keys kg..kg+31: two 16x16 C tiles, K-dim = d_head
  v8f c0 = {}, c1 = {};
  const _Float16* p0 = Kh + (size_t)(kg + m) * D_HEAD + hi * 8;
  const _Float16* p1 = p0 + 16 * D_HEAD;
  c0 = wmma_f16(aq0, ldB_f16(p0), c0);
  c1 = wmma_f16(aq0, ldB_f16(p1), c1);
  c0 = wmma_f16(aq1, ldB_f16(p0 + 32), c0);
  c1 = wmma_f16(aq1, ldB_f16(p1 + 32), c1);

  // C-layout raw scores -> LDS (f32 row-major 16x32)
  #pragma unroll
  for (int r = 0; r < 8; ++r) {
    sl[(r + hi * 8) * 32 + m]      = c0[r];
    sl[(r + hi * 8) * 32 + 16 + m] = c1[r];
  }
  __builtin_amdgcn_wave_barrier();
  asm volatile("s_wait_dscnt 0" ::: "memory");   // LDS is in-order per wave

  // reload own A-layout row slice: row = m, K = hi*8 + {0..7, 16..23}
  const float* srow = sl + m * 32 + hi * 8;
  const float4 f0 = ((const float4*)srow)[0];
  const float4 f1 = ((const float4*)srow)[1];
  const float4 f2 = ((const float4*)srow)[4];
  const float4 f3 = ((const float4*)srow)[5];
  float sv[16] = {f0.x, f0.y, f0.z, f0.w, f1.x, f1.y, f1.z, f1.w,
                  f2.x, f2.y, f2.z, f2.w, f3.x, f3.y, f3.z, f3.w};
  __builtin_amdgcn_wave_barrier();

  if (MASKED) {
    const int row = qrow0 + m;
    #pragma unroll
    for (int j = 0; j < 16; ++j) {
      const int key = kg + hi * 8 + (j & 7) + ((j >> 3) << 4);
      sv[j] = (key <= row) ? sv[j] : -INFINITY;
    }
  }

  // row max: local tree + one cross-half merge
  float bm = sv[0];
  #pragma unroll
  for (int j = 1; j < 16; ++j) bm = fmaxf(bm, sv[j]);
  bm = fmaxf(bm, __shfl_xor(bm, 16, 32));
  const float mnew = fmaxf(mrow, bm);
  const float corr = __expf(mrow - mnew);
  mrow = mnew;

  // P fragment built directly in A-layout
  v16h ap;
  #pragma unroll
  for (int j = 0; j < 16; ++j) ap[j] = (_Float16)__expf(sv[j] - mnew);

  // broadcast per-row corr to C-layout rows hi*8..hi*8+7 (contiguous)
  sl[512 + m] = corr;                  // lanes m and m+16 write identical value
  __builtin_amdgcn_wave_barrier();
  asm volatile("s_wait_dscnt 0" ::: "memory");
  const float4 cA = ((const float4*)(sl + 512 + hi * 8))[0];
  const float4 cB = ((const float4*)(sl + 512 + hi * 8))[1];
  const float corrs[8] = {cA.x, cA.y, cA.z, cA.w, cB.x, cB.y, cB.z, cB.w};
  __builtin_amdgcn_wave_barrier();

  #pragma unroll
  for (int r = 0; r < 8; ++r) {
    lacc[r]    *= corrs[r];
    oacc[0][r] *= corrs[r]; oacc[1][r] *= corrs[r];
    oacc[2][r] *= corrs[r]; oacc[3][r] *= corrs[r];
  }

  // row-sum of P via matrix unit: lacc += P * ones
  lacc = wmma_f16(ap, bones, lacc);

  // O += P(16x32) @ V(32x64); V^T layout makes each B-frag 2 vector loads
  #pragma unroll
  for (int dt = 0; dt < 4; ++dt) {
    const _Float16* pv = Vth + (size_t)(dt * 16 + m) * SEQ + kg + hi * 8;
    oacc[dt] = wmma_f16(ap, ldB_f16(pv), oacc[dt]);
  }
}

// ---------------------------------------------------------------------------
// Flash attention, causal. 8 waves/WG; wave w owns q-rows [qrow0, qrow0+16).
// Q layout [BH,S,dh]; K layout [BH,S,dh]; V layout [BH,dh,S] (transposed).
// 1/sqrt(dh) and biases already folded in by proj_qkv.
// ---------------------------------------------------------------------------
__global__ __launch_bounds__(256) void attn_flash(const _Float16* __restrict__ Q,
                                                  const _Float16* __restrict__ K,
                                                  const _Float16* __restrict__ Vt,
                                                  _Float16* __restrict__ Outp) {
  __shared__ __align__(16) float slds[8][16 * 32 + 16]; // scores + corr, per wave

  const int lane = threadIdx.x & 31;
  const int wave = threadIdx.x >> 5;
  const int bh   = blockIdx.x >> 4;                     // 0..63
  const int qblk = blockIdx.x & 15;
  const int b = bh >> 4, h = bh & 15;
  const _Float16* Qh  = Q  + (size_t)bh * (SEQ * D_HEAD);
  const _Float16* Kh  = K  + (size_t)bh * (SEQ * D_HEAD);
  const _Float16* Vth = Vt + (size_t)bh * (D_HEAD * SEQ);
  const int qrow0 = qblk * 128 + wave * 16;
  const int m  = lane & 15;
  const int hi = lane >> 4;

  // Q tile as two A-fragments (dh 0..31 and 32..63), resident all loop long
  const _Float16* pq = Qh + (size_t)(qrow0 + m) * D_HEAD + hi * 8;
  const v16h aq0 = ldA_f16(pq);
  const v16h aq1 = ldA_f16(pq + 32);

  v16h bones;                                   // all-ones B fragment
  #pragma unroll
  for (int j = 0; j < 16; ++j) bones[j] = (_Float16)1.0f;

  float mrow = -INFINITY;                       // per-lane row state (row = m)
  v8f oacc[4] = {};
  v8f lacc = {};

  float* sl = &slds[wave][0];
  const int kblocks = (qrow0 + 15) / 32 + 1;    // causal trip count (per wave)

  // interior blocks: provably unmasked (max key < qrow0)
  for (int t = 0; t < kblocks - 1; ++t)
    attn_block<false>(t * 32, qrow0, m, hi, Kh, Vth, sl, aq0, aq1, bones,
                      mrow, lacc, oacc);
  // single diagonal block with causal masking
  attn_block<true>((kblocks - 1) * 32, qrow0, m, hi, Kh, Vth, sl, aq0, aq1, bones,
                   mrow, lacc, oacc);

  // normalize (lacc holds row-sum in every column) and write [B, S, H*dh] f16
  #pragma unroll
  for (int r = 0; r < 8; ++r) {
    const float inv = 1.0f / lacc[r];
    const int s = qrow0 + r + hi * 8;
    const size_t base = ((size_t)b * SEQ + s) * D_MODEL + h * D_HEAD + m;
    #pragma unroll
    for (int dt = 0; dt < 4; ++dt)
      Outp[base + dt * 16] = (_Float16)(oacc[dt][r] * inv);
  }
}

// ---------------------------------------------------------------------------
// Output projection: d_out[m,n] = sum_k attn16[m,k]*wo16[n,k] + bo[n]  (f32)
// Same 32x64-per-wave tiling as proj_qkv, A already f16.
// ---------------------------------------------------------------------------
__global__ __launch_bounds__(128) void out_proj(const _Float16* __restrict__ A,
                                                const _Float16* __restrict__ W,
                                                const float* __restrict__ bias,
                                                float* __restrict__ Out) {
  const int lane = threadIdx.x & 31;
  const int wave = threadIdx.x >> 5;
  const int tile = blockIdx.x * 4 + wave;        // 0..4095
  const int tm   = tile >> 4;                    // 0..255
  const int tng  = tile & 15;                    // 0..15
  const int m    = lane & 15;
  const int hi   = lane >> 4;

  const _Float16* arow0 = A + (size_t)(tm * 32 + m) * D_MODEL + hi * 8;
  const _Float16* arow1 = arow0 + (size_t)16 * D_MODEL;
  const _Float16* brow  = W + (size_t)(tng * 64 + m) * D_MODEL + hi * 8;

  v8f acc[2][4] = {};
  #pragma unroll 2
  for (int k0 = 0; k0 < D_MODEL; k0 += 32) {
    const v16h a0 = ldA_f16(arow0 + k0);
    const v16h a1 = ldA_f16(arow1 + k0);
    #pragma unroll
    for (int nt = 0; nt < 4; ++nt) {
      const v16h b = ldB_f16(brow + (size_t)nt * 16 * D_MODEL + k0);
      acc[0][nt] = wmma_f16(a0, b, acc[0][nt]);
      acc[1][nt] = wmma_f16(a1, b, acc[1][nt]);
    }
  }

  #pragma unroll
  for (int nt = 0; nt < 4; ++nt) {
    const int n_glob = tng * 64 + nt * 16 + m;
    const float bb = bias[n_glob];
    #pragma unroll
    for (int mi = 0; mi < 2; ++mi) {
      #pragma unroll
      for (int r = 0; r < 8; ++r)
        Out[(size_t)(tm * 32 + mi * 16 + r + hi * 8) * D_MODEL + n_glob] =
            acc[mi][nt][r] + bb;
    }
  }
}

// ---------------------------------------------------------------------------
extern "C" void kernel_launch(void* const* d_in, const int* in_sizes, int n_in,
                              void* d_out, int out_size, void* d_ws, size_t ws_size,
                              hipStream_t stream) {
  (void)in_sizes; (void)n_in; (void)out_size; (void)ws_size;
  const float* xq = (const float*)d_in[0];
  const float* xk = (const float*)d_in[1];
  const float* xv = (const float*)d_in[2];
  // d_in[3] = causal mask (int32) — recomputed on the fly, never read
  const float* wq = (const float*)d_in[4];
  const float* wk = (const float*)d_in[5];
  const float* wv = (const float*)d_in[6];
  const float* bq = (const float*)d_in[7];
  const float* bk = (const float*)d_in[8];
  const float* bv = (const float*)d_in[9];
  const float* wo = (const float*)d_in[10];
  const float* bo = (const float*)d_in[11];
  float* out = (float*)d_out;

  const size_t WELEMS = (size_t)D_MODEL * D_MODEL;              // 1M halfs / weight
  const size_t HELEMS = (size_t)BATCH * N_HEADS * SEQ * D_HEAD; // 8M halfs
  _Float16* wq16 = (_Float16*)d_ws;
  _Float16* wk16 = wq16 + WELEMS;
  _Float16* wv16 = wk16 + WELEMS;
  _Float16* wo16 = wv16 + WELEMS;
  _Float16* q16  = wo16 + WELEMS;     // [B,H,S,dh]
  _Float16* k16  = q16 + HELEMS;      // [B,H,S,dh]
  _Float16* vt16 = k16 + HELEMS;      // [B,H,dh,S]  (transposed V)
  _Float16* a16  = vt16 + HELEMS;     // [B,S,D] attn result

  // 1) downcast weights (1M elems each, 4 halfs/thread)
  cvt_f32_f16<<<1024, 256, 0, stream>>>(wq, wq16);
  cvt_f32_f16<<<1024, 256, 0, stream>>>(wk, wk16);
  cvt_f32_f16<<<1024, 256, 0, stream>>>(wv, wv16);
  cvt_f32_f16<<<1024, 256, 0, stream>>>(wo, wo16);

  // 2) QKV projections: 4096 32x64 tiles, 4 waves/block. 1/sqrt(dh) folded in Q.
  proj_qkv<false><<<1024, 128, 0, stream>>>(xq, wq16, bq, q16, 0.125f);
  proj_qkv<false><<<1024, 128, 0, stream>>>(xk, wk16, bk, k16, 1.0f);
  proj_qkv<true ><<<1024, 128, 0, stream>>>(xv, wv16, bv, vt16, 1.0f);

  // 3) causal flash attention: B*H*16 blocks, 8 waves each (128 q-rows/block)
  attn_flash<<<BATCH * N_HEADS * 16, 256, 0, stream>>>(q16, k16, vt16, a16);

  // 4) output projection to f32
  out_proj<<<1024, 128, 0, stream>>>(a16, wo16, bo, out);
}
